// BatchTreeEncoder_77807627534867
// MI455X (gfx1250) — matrix operand order repaired and checked by reading, not compiled, not run on hardware
//
#include <hip/hip_runtime.h>

typedef __attribute__((ext_vector_type(16))) __bf16 v16bf;
typedef __attribute__((ext_vector_type(8)))  float  v8f;

#define ENC 128
#define BSZ 256
#define G3  384   // 3*ENC

// ---- helpers ---------------------------------------------------------------
// f32 -> bf16 (RNE): let the backend pick the native gfx1250 conversion op.
__device__ __forceinline__ unsigned short f2bf(float f) {
    __bf16 b = (__bf16)f;
    return __builtin_bit_cast(unsigned short, b);
}
__device__ __forceinline__ float bf2f(unsigned short b) {
    return __uint_as_float(((unsigned)b) << 16);
}
// monotone float -> u32 key for atomic max over signed floats
__device__ __forceinline__ unsigned fkey(float x) {
    unsigned u = __float_as_uint(x);
    return (u & 0x80000000u) ? ~u : (u | 0x80000000u);
}
__device__ __forceinline__ float unfkey(unsigned u) {
    return __uint_as_float((u & 0x80000000u) ? (u & 0x7FFFFFFFu) : ~u);
}
// CDNA5 hardware V_TANH_F32 (gfx1250); branch-free fallback otherwise.
__device__ __forceinline__ float fast_tanh(float x) {
#if __has_builtin(__builtin_amdgcn_tanhf)
    return __builtin_amdgcn_tanhf(x);
#else
    float t = __expf(-2.0f * fabsf(x));
    float r = (1.0f - t) / (1.0f + t);
    return copysignf(r, x);
#endif
}
__device__ __forceinline__ float fast_sigmoid(float x) {
    return 0.5f * fast_tanh(0.5f * x) + 0.5f;
}

union AFrag { v16bf v; unsigned u[8]; };

// A fragment (16x32 bf16) from bf16 LDS tile, row stride S (elements), K-base kb.
__device__ __forceinline__ v16bf load_a_lds_bf16(const unsigned short* buf, int S, int kb, int lane) {
    AFrag a;
    int m = lane & 15, h = lane >> 4;
    const unsigned short* p = buf + m * S + kb;
#pragma unroll
    for (int j = 0; j < 4; ++j)
        a.u[j] = *(const unsigned*)(p + 8 * h + 2 * j);
#pragma unroll
    for (int j = 0; j < 4; ++j)
        a.u[4 + j] = *(const unsigned*)(p + 16 + 8 * h + 2 * j);
    return a.v;
}
// A fragment from f32 LDS tile (convert on the fly).
__device__ __forceinline__ v16bf load_a_lds_f32(const float* buf, int S, int kb, int lane) {
    AFrag a;
    int m = lane & 15, h = lane >> 4;
    const float* p = buf + m * S + kb;
#pragma unroll
    for (int j = 0; j < 8; ++j) {
        int k = (j < 4) ? (8 * h + 2 * j) : (16 + 8 * h + 2 * (j - 4));
        unsigned lo = f2bf(p[k]);
        unsigned hi = f2bf(p[k + 1]);
        a.u[j] = lo | (hi << 16);
    }
    return a.v;
}
// B fragment (32x16 bf16): element(K,N) = W[N*128 + K]; per lane K = 8h+0..7, 16+8h+0..7.
__device__ __forceinline__ v16bf load_b_glob(const unsigned short* __restrict__ W, int n0, int kb, int lane) {
    AFrag b;
    int n = n0 + (lane & 15), h = lane >> 4;
    const unsigned short* p = W + n * ENC + kb;
#pragma unroll
    for (int j = 0; j < 8; ++j)
        b.u[j] = *(const unsigned*)(p + ((j < 4) ? (8 * h + 2 * j) : (16 + 8 * h + 2 * (j - 4))));
    return b.v;
}

// ---- LDS layout (attention region overlaps GRU region) ---------------------
#define SX   132          // bf16 row stride (128 + 4 pad)
#define SH0  132          // f32 row stride
#define SG   388          // f32 row stride for 384-wide gate buffers
#define OFF_X   0                       // 16*132*2       = 4224
#define OFF_H0  4224                    // 16*132*4       = 8448
#define OFF_SC  12672                   // 64*4
#define OFF_AL  12928                   // 64*4
#define OFF_B   13184
#define SMEM_BYTES (OFF_B + 2 * 16 * SG * 4)   // 13184 + 49664 = 62848

__global__ __launch_bounds__(256) void level_kernel(
    const int*   __restrict__ tokens_lvl,      // (n_d, 256)
    const float* __restrict__ emb,             // (50000,128)
    const unsigned short* __restrict__ Wih,    // 384x128 bf16, K-contig
    const unsigned short* __restrict__ Whh,    // 384x128 bf16, K-contig
    const unsigned short* __restrict__ WsentT, // 128x128 bf16 (N-major, K-contig)
    const float* __restrict__ b_ih,            // 384
    const float* __restrict__ b_hh,            // 384
    const float* __restrict__ sent_bias,       // 128
    const float* __restrict__ ctx,             // 128
    const unsigned short* __restrict__ prev_h, // (4*n_d,256,128) bf16 (null if leaf)
    unsigned short* __restrict__ cur_h,        // (n_d,256,128) bf16
    unsigned* __restrict__ maxkey,             // 256x128 keys
    int leaf)
{
    __shared__ __align__(16) char smem[SMEM_BYTES];
    unsigned short* lds_x  = (unsigned short*)(smem + OFF_X);
    float*          lds_h0 = (float*)(smem + OFF_H0);
    float*          lds_sc = (float*)(smem + OFF_SC);
    float*          lds_al = (float*)(smem + OFF_AL);
    unsigned short* lds_ch = (unsigned short*)(smem + OFF_B);               // attn phase
    unsigned short* lds_u  = (unsigned short*)(smem + OFF_B + 4*16*SX*2);   // attn phase
    float*          lds_gi = (float*)(smem + OFF_B);                        // gru phase
    float*          lds_gh = (float*)(smem + OFF_B + 16 * SG * 4);          // gru phase

    const int tid  = threadIdx.x;
    const int lane = tid & 31;
    const int wave = tid >> 5;
    const int node = blockIdx.x >> 4;
    const int b0   = (blockIdx.x & 15) * 16;

    // ---- gather x = emb[tokens], convert to bf16 ----
    {
        int m  = tid >> 4;
        int e0 = (tid & 15) * 8;
        int tok = tokens_lvl[node * BSZ + b0 + m];
        const float* er = emb + (long)tok * ENC + e0;
        unsigned short* xr = lds_x + m * SX + e0;
#pragma unroll
        for (int j = 0; j < 8; ++j) xr[j] = f2bf(er[j]);
    }

    if (!leaf) {
        // ---- load 4 children (bf16) ----
        {
            int m  = tid >> 4;
            int e0 = (tid & 15) * 8;
#pragma unroll
            for (int k = 0; k < 4; ++k) {
                const unsigned short* hr =
                    prev_h + ((long)(4 * node + k) * BSZ + b0 + m) * ENC + e0;
                unsigned short* cr = lds_ch + (k * 16 + m) * SX + e0;
#pragma unroll
                for (int j = 0; j < 4; ++j)
                    ((unsigned*)cr)[j] = ((const unsigned*)hr)[j];
            }
        }
        __syncthreads();
        // ---- u = tanh(ch @ Wsent + b): 32 WMMA tiles over 8 waves ----
        for (int t = wave; t < 32; t += 8) {
            int k = t >> 3, n0 = (t & 7) * 16;
            v8f c = {};
#pragma unroll
            for (int s = 0; s < 4; ++s) {
                v16bf a = load_a_lds_bf16(lds_ch + k * 16 * SX, SX, 32 * s, lane);
                v16bf b = load_b_glob(WsentT, n0, 32 * s, lane);
                c = __builtin_amdgcn_wmma_f32_16x16x32_bf16(false, a, false, b,
                                                            (short)0, c, false, false);
            }
            int n = lane & 15, h = lane >> 4;
            float bias = sent_bias[n0 + n];
#pragma unroll
            for (int r = 0; r < 8; ++r) {
                int m = r + 8 * h;
                lds_u[(k * 16 + m) * SX + n0 + n] = f2bf(fast_tanh(c[r] + bias));
            }
        }
        __syncthreads();
        // ---- scores = tanh(u . ctx) ----
        if (tid < 64) {
            int k = tid >> 4, m = tid & 15;
            const unsigned short* ur = lds_u + (k * 16 + m) * SX;
            float s = 0.f;
            for (int e = 0; e < ENC; ++e) s += bf2f(ur[e]) * ctx[e];
            lds_sc[k * 16 + m] = fast_tanh(s);
        }
        __syncthreads();
        // ---- softmax over K=4 children ----
        if (tid < 16) {
            int m = tid;
            float s0 = lds_sc[m], s1 = lds_sc[16 + m], s2 = lds_sc[32 + m], s3 = lds_sc[48 + m];
            float mx = fmaxf(fmaxf(s0, s1), fmaxf(s2, s3));
            float e0 = __expf(s0 - mx), e1 = __expf(s1 - mx),
                  e2 = __expf(s2 - mx), e3 = __expf(s3 - mx);
            float inv = 1.f / (e0 + e1 + e2 + e3);
            lds_al[m] = e0 * inv; lds_al[16 + m] = e1 * inv;
            lds_al[32 + m] = e2 * inv; lds_al[48 + m] = e3 * inv;
        }
        __syncthreads();
        // ---- h0 = sum_k alpha_k * ch_k (f32) ----
        {
            int m = tid >> 4, e0 = (tid & 15) * 8;
            float a0 = lds_al[m], a1 = lds_al[16 + m], a2 = lds_al[32 + m], a3 = lds_al[48 + m];
#pragma unroll
            for (int j = 0; j < 8; ++j) {
                int e = e0 + j;
                lds_h0[m * SH0 + e] =
                      a0 * bf2f(lds_ch[(m)      * SX + e])
                    + a1 * bf2f(lds_ch[(16 + m) * SX + e])
                    + a2 * bf2f(lds_ch[(32 + m) * SX + e])
                    + a3 * bf2f(lds_ch[(48 + m) * SX + e]);
            }
        }
    } else {
        // leaf: h0 = 0, gh = b_hh broadcast (skip the h0 GEMM entirely)
        int m = tid >> 4, e0 = (tid & 15) * 8;
#pragma unroll
        for (int j = 0; j < 8; ++j) lds_h0[m * SH0 + e0 + j] = 0.f;
        for (int i = tid; i < 16 * G3; i += 256)
            lds_gh[(i / G3) * SG + (i % G3)] = b_hh[i % G3];
    }
    __syncthreads();

    // ---- GRU gates: gi (24 tiles) + gh (24 tiles, skipped for leaves) ----
    {
        int ntiles = leaf ? 24 : 48;
        for (int t = wave; t < ntiles; t += 8) {
            bool is_gi = t < 24;
            int  jt = is_gi ? t : t - 24;
            int  n0 = jt * 16;
            const unsigned short* W = is_gi ? Wih : Whh;
            v8f c = {};
#pragma unroll
            for (int s = 0; s < 4; ++s) {
                v16bf a = is_gi ? load_a_lds_bf16(lds_x, SX, 32 * s, lane)
                                : load_a_lds_f32(lds_h0, SH0, 32 * s, lane);
                v16bf b = load_b_glob(W, n0, 32 * s, lane);
                c = __builtin_amdgcn_wmma_f32_16x16x32_bf16(false, a, false, b,
                                                            (short)0, c, false, false);
            }
            int n = lane & 15, h = lane >> 4;
            float bias = is_gi ? b_ih[n0 + n] : b_hh[n0 + n];
            float* dst = is_gi ? lds_gi : lds_gh;
#pragma unroll
            for (int r = 0; r < 8; ++r)
                dst[(r + 8 * h) * SG + n0 + n] = c[r] + bias;
        }
    }
    __syncthreads();

    // ---- elementwise gates, store hn (bf16) + atomic max-pool ----
    {
        int m = tid >> 4, e0 = (tid & 15) * 8;
        long base = ((long)node * BSZ + b0 + m) * ENC;
#pragma unroll
        for (int j = 0; j < 8; ++j) {
            int e = e0 + j;
            float r  = fast_sigmoid(lds_gi[m * SG + e]       + lds_gh[m * SG + e]);
            float z  = fast_sigmoid(lds_gi[m * SG + 128 + e] + lds_gh[m * SG + 128 + e]);
            float nn = fast_tanh(lds_gi[m * SG + 256 + e] + r * lds_gh[m * SG + 256 + e]);
            float hn = (1.f - z) * nn + z * lds_h0[m * SH0 + e];
            cur_h[base + e] = f2bf(hn);
            atomicMax(&maxkey[(b0 + m) * ENC + e], fkey(hn));
        }
    }
}

// ---- prep: weights f32->bf16 (W_sent transposed to N-major), init max keys --
__global__ __launch_bounds__(256) void prep_kernel(
    const float* __restrict__ Wih, const float* __restrict__ Whh,
    const float* __restrict__ sentW,
    unsigned short* __restrict__ wih_bf, unsigned short* __restrict__ whh_bf,
    unsigned short* __restrict__ wsentT, unsigned* __restrict__ mkey)
{
    int i = blockIdx.x * 256 + threadIdx.x;
    if (i < 49152) {
        wih_bf[i] = f2bf(Wih[i]);
    } else if (i < 98304) {
        int j = i - 49152;
        whh_bf[j] = f2bf(Whh[j]);
    } else if (i < 114688) {
        int j = i - 98304, f = j >> 7, e = j & 127;
        wsentT[f * 128 + e] = f2bf(sentW[e * 128 + f]);
    } else if (i < 147456) {
        mkey[i - 114688] = 0u;
    }
}

__global__ __launch_bounds__(256) void final_kernel(const unsigned* __restrict__ mkey,
                                                    float* __restrict__ out)
{
    int i = blockIdx.x * 256 + threadIdx.x;
    if (i < BSZ * ENC) out[i] = unfkey(mkey[i]);
}

// ---- host launch ------------------------------------------------------------
extern "C" void kernel_launch(void* const* d_in, const int* in_sizes, int n_in,
                              void* d_out, int out_size, void* d_ws, size_t ws_size,
                              hipStream_t stream)
{
    const int*   tokens = (const int*)  d_in[0];
    const float* emb    = (const float*)d_in[1];
    const float* W_ih   = (const float*)d_in[2];
    const float* W_hh   = (const float*)d_in[3];
    const float* b_ih   = (const float*)d_in[4];
    const float* b_hh   = (const float*)d_in[5];
    const float* sentW  = (const float*)d_in[6];
    const float* sentB  = (const float*)d_in[7];
    const float* ctx    = (const float*)d_in[8];

    char* ws = (char*)d_ws;
    unsigned short* wih_bf = (unsigned short*)(ws + 0);
    unsigned short* whh_bf = (unsigned short*)(ws + 98304);
    unsigned short* wsentT = (unsigned short*)(ws + 196608);
    unsigned*       mkey   = (unsigned*)(ws + 229376);          // 256*128 keys
    unsigned short* hA     = (unsigned short*)(ws + 360448);    // up to 1024 nodes
    unsigned short* hB     = (unsigned short*)(ws + 360448 + (size_t)1024*256*128*2);

    prep_kernel<<<576, 256, 0, stream>>>(W_ih, W_hh, sentW, wih_bf, whh_bf, wsentT, mkey);

    // levels, leaves -> root:  (n_d, row start in tokens)
    const int nd[6]    = {1024, 256, 64, 16, 4, 1};
    const int start[6] = {341, 85, 21, 5, 1, 0};
    unsigned short* bufs[2] = {hA, hB};
    const unsigned short* prev = nullptr;
    for (int i = 0; i < 6; ++i) {
        unsigned short* cur = bufs[i & 1];
        int leaf = (i == 0);
        level_kernel<<<nd[i] * 16, 256, 0, stream>>>(
            tokens + (long)start[i] * BSZ, emb,
            wih_bf, whh_bf, wsentT, b_ih, b_hh, sentB, ctx,
            leaf ? nullptr : prev, cur, mkey, leaf);
        prev = cur;
    }
    final_kernel<<<128, 256, 0, stream>>>(mkey, (float*)d_out);
}
